// FlashAttentionOp_89953795048097
// MI455X (gfx1250) — matrix operand order repaired and checked by reading, not compiled, not run on hardware
//
#include <hip/hip_runtime.h>

// CDNA5 / gfx1250 flash attention forward.
// D = A*B + C via v_wmma_f32_16x16x32_f16 (wave32).
// 256-thread blocks = 8 waves; each wave owns a 16-row Q strip (128 q rows/block)
// sharing one K/V LDS tile -> halves staging work vs 4-wave blocks.
// K+V (128 MB fp32) fit in MI455X's 192 MB L2, so cross-block tile re-reads
// are L2-resident; HBM traffic stays at the compulsory ~256 MB.

typedef _Float16 v16h __attribute__((ext_vector_type(16)));
typedef _Float16 h2   __attribute__((ext_vector_type(2)));
typedef float    v8f  __attribute__((ext_vector_type(8)));

#define NEG_INF (-1e30f)

// v_cvt_pk_rtz_f16_f32: pack two f32 -> two f16 in one VALU op.
static __device__ __forceinline__ h2 cvt_pk(float x, float y) {
    return __builtin_bit_cast(h2, __builtin_amdgcn_cvt_pkrtz(x, y));
}

constexpr int Bc = 4;      // batch
constexpr int Sc = 2048;   // sequence
constexpr int Hc = 16;     // heads
constexpr int Dc = 128;    // head dim
constexpr int QT = 128;    // q rows per block (8 waves x 16)
constexpr int KT = 32;     // keys per LDS tile
constexpr int KSTR = 138;  // f16 stride of K tile rows (pad: conflict-free B reads)
constexpr int VSTR = 34;   // f16 stride of V^T rows (per d)
constexpr int PSTR = 34;   // f16 stride of per-wave P strip rows

__launch_bounds__(256, 1)
__global__ void fa_fwd_gfx1250(const float* __restrict__ q,
                               const float* __restrict__ k,
                               const float* __restrict__ v,
                               const int*   __restrict__ seg,
                               float* __restrict__ out)
{
    __shared__ __align__(16) _Float16 sK [KT * KSTR];    // K tile, [key][d]
    __shared__ __align__(16) _Float16 sVT[Dc * VSTR];    // V tile transposed, [d][key]
    __shared__ __align__(16) _Float16 sP [8][16 * PSTR]; // per-wave P strip, [qrow][key]

    const int tid  = threadIdx.x;
    const int lane = tid & 31;
    const int wid  = tid >> 5;
    const int n    = lane & 15;   // N-index / A-row index within half
    const int hi   = lane >> 4;   // half-wave select

    const int qt = blockIdx.x;
    const int h  = blockIdx.y;
    const int b  = blockIdx.z;

    const int blockQ = qt * QT;
    const int wq     = blockQ + wid * 16;   // this wave's 16-row q base

    // ---- Q -> 4 A-fragments (16x32 f16 each), ISA 16-bit A layout ----
    v16h aQ[4];
    {
        const int qrow = wq + n;                    // A layout: lane holds row M = lane%16
        const float* qp = q + (((size_t)b * Sc + qrow) * Hc + h) * Dc;
        #pragma unroll
        for (int kd = 0; kd < 4; ++kd) {
            #pragma unroll
            for (int r = 0; r < 8; ++r) {
                // K(lane,r) = {0..7,16..23} + 8*hi pattern, pairs (2r,2r+1)
                int d = kd * 32 + ((r & 4) ? 16 : 0) + 2 * (r & 3) + 8 * hi;
                float2 f = *(const float2*)(qp + d);
                h2 p = cvt_pk(f.x, f.y);
                aQ[kd][2 * r]     = p[0];
                aQ[kd][2 * r + 1] = p[1];
            }
        }
    }

    // per-row online-softmax state; this lane covers rows M = r + 8*hi (C layout)
    float mrow[8], lrow[8];
    int   segq[8], qg[8];
    #pragma unroll
    for (int r = 0; r < 8; ++r) {
        mrow[r] = NEG_INF;
        lrow[r] = 0.0f;
        qg[r]   = wq + r + 8 * hi;
        segq[r] = seg[(size_t)b * Sc + qg[r]];
    }

    v8f acc[8] = {};   // 16x128 f32 output accumulator (8 chunks of 16 cols)

    const int ktiles = blockQ / KT + QT / KT;   // causal: keys <= blockQ + QT - 1

    for (int kt = 0; kt < ktiles; ++kt) {
        const int k0 = kt * KT;
        __syncthreads();
        // ---- cooperative staging (256 threads): K row-major f16, V^T f16 ----
        {
            const float* kp = k + (((size_t)b * Sc + k0) * Hc + h) * Dc;
            #pragma unroll 4
            for (int j = 0; j < 8; ++j) {
                int pidx = j * 256 + tid;           // 2048 f32-pairs total
                int row  = pidx >> 6;               // 64 pairs per key row
                int dp   = (pidx & 63) * 2;
                float2 f = *(const float2*)(kp + (size_t)row * Hc * Dc + dp);
                *(h2*)&sK[row * KSTR + dp] = cvt_pk(f.x, f.y);
            }
            const float* vp = v + (((size_t)b * Sc + k0) * Hc + h) * Dc;
            #pragma unroll 4
            for (int j = 0; j < 8; ++j) {
                int pidx = j * 256 + tid;
                int row  = pidx >> 6;
                int dp   = (pidx & 63) * 2;
                float2 f = *(const float2*)(vp + (size_t)row * Hc * Dc + dp);
                h2 p = cvt_pk(f.x, f.y);
                sVT[dp * VSTR + row]       = p[0];   // transpose on store
                sVT[(dp + 1) * VSTR + row] = p[1];
            }
        }
        if (kt + 1 < ktiles) {   // global_prefetch_b8 for next K tile
            const float* kn = k + (((size_t)b * Sc + k0 + KT) * Hc + h) * Dc;
            __builtin_prefetch(kn + tid * 16, 0, 0);
        }
        __syncthreads();

        // Wave-uniform causal skip: this wave's rows end at wq+15.
        // (k0, wq are wave-uniform -> EXEC stays all-ones inside, as WMMA requires.)
        if (k0 > wq + 15) continue;
        const bool needCausal = (k0 + KT - 1 > wq);   // diagonal-boundary tile?

        // ---- scores = Q * K^T : two 16-key column groups ----
        v8f sc[2];
        #pragma unroll
        for (int g = 0; g < 2; ++g) {
            v8f c = {};
            #pragma unroll
            for (int kd = 0; kd < 4; ++kd) {
                v16h bK;   // B[dcontr, key] = K[key g*16+n][d]; pairs contiguous in d
                const _Float16* base = &sK[(g * 16 + n) * KSTR + kd * 32];
                #pragma unroll
                for (int r = 0; r < 8; ++r) {
                    h2 p = *(const h2*)(base + ((r & 4) ? 16 : 0) + 2 * (r & 3) + 8 * hi);
                    bK[2 * r] = p[0]; bK[2 * r + 1] = p[1];
                }
                c = __builtin_amdgcn_wmma_f32_16x16x32_f16(
                        false, aQ[kd], false, bK, (short)0, c, false, false);
            }
            sc[g] = c;
        }

        // ---- segment (+ causal on boundary tiles) mask, online softmax ----
        const int segk0 = seg[(size_t)b * Sc + k0 + n];
        const int segk1 = seg[(size_t)b * Sc + k0 + 16 + n];
        float p0[8], p1[8], alpha[8];
        #pragma unroll
        for (int r = 0; r < 8; ++r) {
            const int kg0 = k0 + n, kg1 = k0 + 16 + n;
            bool ok0 = (segk0 == segq[r]) && (!needCausal || (kg0 <= qg[r]));
            bool ok1 = (segk1 == segq[r]) && (!needCausal || (kg1 <= qg[r]));
            float s0 = ok0 ? sc[0][r] : NEG_INF;
            float s1 = ok1 ? sc[1][r] : NEG_INF;
            float t = fmaxf(s0, s1);
            #pragma unroll
            for (int msk = 1; msk < 16; msk <<= 1)
                t = fmaxf(t, __shfl_xor(t, msk, 16));   // rows live in 16-lane halves
            float mnew = fmaxf(mrow[r], t);
            float a    = __expf(mrow[r] - mnew);
            mrow[r]  = mnew;
            alpha[r] = a;
            p0[r] = __expf(s0 - mnew);
            p1[r] = __expf(s1 - mnew);
            float ps = p0[r] + p1[r];
            #pragma unroll
            for (int msk = 1; msk < 16; msk <<= 1)
                ps += __shfl_xor(ps, msk, 16);
            lrow[r] = lrow[r] * a + ps;
        }
        #pragma unroll
        for (int c = 0; c < 8; ++c)
            #pragma unroll
            for (int r = 0; r < 8; ++r)
                acc[c][r] *= alpha[r];

        // ---- C-layout P -> f16 LDS strip -> A-layout fragment ----
        _Float16* sp = sP[wid];
        #pragma unroll
        for (int r = 0; r < 8; ++r) {
            sp[(r + 8 * hi) * PSTR + n]      = (_Float16)p0[r];
            sp[(r + 8 * hi) * PSTR + 16 + n] = (_Float16)p1[r];
        }
        v16h aP;
        {
            const _Float16* base = &sp[n * PSTR];
            #pragma unroll
            for (int r = 0; r < 8; ++r) {
                int kk = ((r & 4) ? 16 : 0) + 2 * (r & 3) + 8 * hi;
                h2 p = *(const h2*)(base + kk);
                aP[2 * r] = p[0]; aP[2 * r + 1] = p[1];
            }
        }

        // ---- acc += P * V ----
        #pragma unroll
        for (int c = 0; c < 8; ++c) {
            v16h bV;   // B[key, d] from V^T: pairs contiguous in key
            const _Float16* base = &sVT[(c * 16 + n) * VSTR];
            #pragma unroll
            for (int r = 0; r < 8; ++r) {
                h2 p = *(const h2*)(base + ((r & 4) ? 16 : 0) + 2 * (r & 3) + 8 * hi);
                bV[2 * r] = p[0]; bV[2 * r + 1] = p[1];
            }
            acc[c] = __builtin_amdgcn_wmma_f32_16x16x32_f16(
                        false, aP, false, bV, (short)0, acc[c], false, false);
        }
    }

    // ---- epilogue: O = acc / l, write [b, q, h, d] ----
    #pragma unroll
    for (int r = 0; r < 8; ++r) {
        float inv = 1.0f / lrow[r];
        float* op = out + (((size_t)b * Sc + qg[r]) * Hc + h) * Dc + n;
        #pragma unroll
        for (int c = 0; c < 8; ++c)
            op[c * 16] = acc[c][r] * inv;
    }
}

extern "C" void kernel_launch(void* const* d_in, const int* in_sizes, int n_in,
                              void* d_out, int out_size, void* d_ws, size_t ws_size,
                              hipStream_t stream) {
    const float* q   = (const float*)d_in[0];
    const float* k   = (const float*)d_in[1];
    const float* v   = (const float*)d_in[2];
    const int*   seg = (const int*)d_in[3];
    float* o = (float*)d_out;

    dim3 grid(Sc / QT, Hc, Bc);   // (16, 16, 4)
    dim3 block(256);              // 8 waves of 32
    hipLaunchKernelGGL(fa_fwd_gfx1250, grid, block, 0, stream, q, k, v, seg, o);
}
